// AMNet_54193897341555
// MI455X (gfx1250) — matrix operand order repaired and checked by reading, not compiled, not run on hardware
//
#include <hip/hip_runtime.h>
#include <hip/hip_bf16.h>

// ---------------------------------------------------------------------------
// AMNet pipeline for MI455X (gfx1250, wave32, WMMA)
//   N=10000, E=640000, IN=512, HID=256, KDEG=5, FN=5
// Dominant cost: res @ res.T (51.2 GF, 400MB f32 out) -> bf16 WMMA SYRK.
// res is produced in bf16 so SYRK fragment loads are pure b128 vector loads
// (no f32->bf16 cvt on the critical path).
// ---------------------------------------------------------------------------

#define N_NODES 10000
#define N_EDGES 640000
#define DIM_IN  512
#define HID     256
#define KDEG    5
#define FN      5
#define NH      (N_NODES * HID)   // 2,560,000

typedef __attribute__((ext_vector_type(16))) __bf16 v16bf;
typedef __attribute__((ext_vector_type(8)))  __bf16 v8bf;
typedef __attribute__((ext_vector_type(8)))  float  v8f;

// Bernstein basis (K=5) power-basis coefficients: coeff[k][i]
__constant__ float d_coeff[36] = {
    1.f,  -5.f,  10.f, -10.f,   5.f,  -1.f,
    0.f,   5.f, -20.f,  30.f, -20.f,   5.f,
    0.f,   0.f,  10.f, -30.f,  30.f, -10.f,
    0.f,   0.f,   0.f,  10.f, -20.f,  10.f,
    0.f,   0.f,   0.f,   0.f,   5.f,  -5.f,
    0.f,   0.f,   0.f,   0.f,   0.f,   1.f
};

// ---------------- WMMA fragment loaders ------------------------------------

// A tile 16x32 (MxK) from row-major f32 source, converting to bf16.
// lane holds row (lane&15); kb=(lane<16)?0:16; k=kb+j contiguous.
__device__ inline v16bf load_a_tile_f32(const float* __restrict__ A, int lda,
                                        int row, int k0, int lane) {
    int kb = (lane & 16) ? 16 : 0;
    const float* p = A + (size_t)row * lda + k0 + kb;
    v16bf a;
#pragma unroll
    for (int j = 0; j < 8; ++j) a[j]     = (__bf16)p[j];
#pragma unroll
    for (int j = 0; j < 8; ++j) a[8 + j] = (__bf16)p[8 + j];
    return a;
}

// B tile 32x16 (KxN) from row-major f32 weights [K, ldb], converting to bf16.
// lane holds column (lane&15); k-half = (lane<16)?0..15:16..31 contiguous.
__device__ inline v16bf load_b_tile_f32(const float* __restrict__ B, int ldb,
                                        int k0, int col, int lane) {
    int kh = (lane & 16) ? 16 : 0;
    const float* p = B + (size_t)(k0 + kh) * ldb + col;
    v16bf b;
#pragma unroll
    for (int j = 0; j < 16; ++j) b[j] = (__bf16)p[(size_t)j * ldb];
    return b;
}

// A tile 16x32 directly from bf16 source: two aligned 16B vector loads.
__device__ inline v16bf load_a_tile_bf(const __bf16* __restrict__ R, int ld,
                                       int row, int k0, int lane) {
    int kb = (lane & 16) ? 16 : 0;
    const __bf16* p = R + (size_t)row * ld + k0 + kb;   // 16B aligned
    v8bf lo = *(const v8bf*)p;
    v8bf hi = *(const v8bf*)(p + 8);
    v16bf a;
#pragma unroll
    for (int j = 0; j < 8; ++j) { a[j] = lo[j]; a[8 + j] = hi[j]; }
    return a;
}

// B tile 32x16 where B[k][n] = R[n][k]: one aligned 32B vector load per lane.
__device__ inline v16bf load_b_tile_bf(const __bf16* __restrict__ R, int ld,
                                       int col, int k0, int lane) {
    int kh = (lane & 16) ? 16 : 0;
    return *(const v16bf*)(R + (size_t)col * ld + k0 + kh); // 32B aligned
}

__device__ inline v8f wmma_bf16(v16bf a, v16bf b, v8f c) {
    return __builtin_amdgcn_wmma_f32_16x16x32_bf16(
        false, a, false, b, (short)0, c, false, false);
}

// ---------------- generic GEMM: C[M,256] = act(A[M,K] @ B[K,256] + bias) ---
// act: 0=none, 1=relu, 2=tanh. grid = (M/16, 2), block = 256 (8 waves).
__global__ __launch_bounds__(256)
void k_gemm256(const float* __restrict__ A, const float* __restrict__ B,
               const float* __restrict__ bias, float* __restrict__ C,
               int M, int K, int act) {
    int lane = threadIdx.x & 31;
    int wave = threadIdx.x >> 5;
    int rowTile = blockIdx.x;
    int colTile = blockIdx.y * 8 + wave;
    int row = rowTile * 16 + (lane & 15);
    int col = colTile * 16 + (lane & 15);
    (void)M;
    v8f c = {};
    for (int k0 = 0; k0 < K; k0 += 32) {
        v16bf a = load_a_tile_f32(A, K, row, k0, lane);
        v16bf b = load_b_tile_f32(B, 256, k0, col, lane);
        c = wmma_bf16(a, b, c);
    }
    int half = (lane >> 4) * 8;
#pragma unroll
    for (int v = 0; v < 8; ++v) {
        int m = rowTile * 16 + half + v;
        float val = c[v] + bias[col];
        if (act == 1)      val = fmaxf(val, 0.f);
        else if (act == 2) val = tanhf(val);
        C[(size_t)m * 256 + col] = val;
    }
}

// ---------------- SYRK: out[m,n] = sum_h res[m,h]*res[n,h] -----------------
// block = 256 (8 waves, 2x4), block tile 64x128, wave tile 32x32 (2x2 WMMA).
// res is bf16 [N,256]; fully L2-resident (5MB), output store (400MB) is the
// HBM floor.
__global__ __launch_bounds__(256)
void k_syrk(const __bf16* __restrict__ R, float* __restrict__ out) {
    int lane = threadIdx.x & 31;
    int wave = threadIdx.x >> 5;
    int wm = wave >> 2;       // 0..1
    int wn = wave & 3;        // 0..3
    int mbase = blockIdx.x * 64 + wm * 32;
    int nbase = blockIdx.y * 128 + wn * 32;
    int r0 = min(mbase + (lane & 15), N_NODES - 1);
    int r1 = min(mbase + 16 + (lane & 15), N_NODES - 1);
    int c0 = min(nbase + (lane & 15), N_NODES - 1);
    int c1 = min(nbase + 16 + (lane & 15), N_NODES - 1);
    v8f a00 = {}, a01 = {}, a10 = {}, a11 = {};
#pragma unroll 2
    for (int k0 = 0; k0 < HID; k0 += 32) {
        v16bf A0 = load_a_tile_bf(R, HID, r0, k0, lane);
        v16bf A1 = load_a_tile_bf(R, HID, r1, k0, lane);
        v16bf B0 = load_b_tile_bf(R, HID, c0, k0, lane);
        v16bf B1 = load_b_tile_bf(R, HID, c1, k0, lane);
        a00 = wmma_bf16(A0, B0, a00);
        a01 = wmma_bf16(A0, B1, a01);
        a10 = wmma_bf16(A1, B0, a10);
        a11 = wmma_bf16(A1, B1, a11);
    }
    int half = (lane >> 4) * 8;
    int ncol0 = nbase + (lane & 15);
    int ncol1 = nbase + 16 + (lane & 15);
#pragma unroll
    for (int v = 0; v < 8; ++v) {
        int m0 = mbase + half + v;
        int m1 = mbase + 16 + half + v;
        if (m0 < N_NODES && ncol0 < N_NODES) out[(size_t)m0 * N_NODES + ncol0] = a00[v];
        if (m0 < N_NODES && ncol1 < N_NODES) out[(size_t)m0 * N_NODES + ncol1] = a01[v];
        if (m1 < N_NODES && ncol0 < N_NODES) out[(size_t)m1 * N_NODES + ncol0] = a10[v];
        if (m1 < N_NODES && ncol1 < N_NODES) out[(size_t)m1 * N_NODES + ncol1] = a11[v];
    }
}

// ---------------- small helper kernels -------------------------------------

__global__ void k_zero(float* __restrict__ p, size_t n) {
    for (size_t i = (size_t)blockIdx.x * blockDim.x + threadIdx.x; i < n;
         i += (size_t)gridDim.x * blockDim.x)
        p[i] = 0.f;
}

__global__ void k_deg(const int* __restrict__ src, float* __restrict__ deg) {
    int e = blockIdx.x * blockDim.x + threadIdx.x;
    if (e < N_EDGES) atomicAdd(&deg[src[e]], 1.0f);
}

__global__ void k_dinv(const float* __restrict__ deg, float* __restrict__ dinv) {
    int i = blockIdx.x * blockDim.x + threadIdx.x;
    if (i < N_NODES) {
        float d = deg[i];
        dinv[i] = d > 0.f ? rsqrtf(d) : 0.f;
    }
}

__global__ void k_scale(const float* __restrict__ cur, float* __restrict__ nxt) {
    for (size_t i = (size_t)blockIdx.x * blockDim.x + threadIdx.x; i < (size_t)NH;
         i += (size_t)gridDim.x * blockDim.x)
        nxt[i] = 0.5f * cur[i];
}

// one block per edge, one thread per hidden channel
__global__ __launch_bounds__(256)
void k_edge(const float* __restrict__ cur, float* __restrict__ nxt,
            const int* __restrict__ src, const int* __restrict__ dst,
            const float* __restrict__ dinv) {
    int e = blockIdx.x;
    int h = threadIdx.x;
    int s = src[e], d = dst[e];
    float w = -0.5f * dinv[s] * dinv[d];
    atomicAdd(&nxt[(size_t)d * HID + h], w * cur[(size_t)s * HID + h]);
}

// hf[n][f][h] += M[f][iPow] * cur[n][h], M[f][i] = sum_k sigmoid(fw[f,k])*coeff[k,i]
__global__ __launch_bounds__(256)
void k_accum(const float* __restrict__ cur, float* __restrict__ hf,
             const float* __restrict__ fw, int iPow) {
    __shared__ float sM[FN];
    if (threadIdx.x < FN) {
        float m = 0.f;
        for (int k = 0; k <= KDEG; ++k) {
            float s = 1.f / (1.f + expf(-fw[threadIdx.x * (KDEG + 1) + k]));
            m += s * d_coeff[k * 6 + iPow];
        }
        sM[threadIdx.x] = m;
    }
    __syncthreads();
    for (size_t idx = (size_t)blockIdx.x * blockDim.x + threadIdx.x; idx < (size_t)NH;
         idx += (size_t)gridDim.x * blockDim.x) {
        size_t n = idx >> 8, h = idx & 255;
        float v = cur[idx];
#pragma unroll
        for (int f = 0; f < FN; ++f)
            hf[((size_t)n * FN + f) * 256 + h] += sM[f] * v;
    }
}

// attention fusion: logits -> softmax -> weighted sum. One block per node.
// Emits res directly in bf16 for the SYRK.
__global__ __launch_bounds__(256)
void k_fuse(const float* __restrict__ hf, const float* __restrict__ hfp,
            const float* __restrict__ xp, __bf16* __restrict__ res) {
    int n = blockIdx.x;
    int t = threadIdx.x;
    __shared__ float sred[256];
    __shared__ float slog[FN];
    __shared__ float ssc[FN];
    float x = xp[(size_t)n * HID + t];
    for (int f = 0; f < FN; ++f) {
        sred[t] = hfp[((size_t)n * FN + f) * HID + t] * x;
        __syncthreads();
        for (int s = 128; s > 0; s >>= 1) {
            if (t < s) sred[t] += sred[t + s];
            __syncthreads();
        }
        if (t == 0) slog[f] = sred[0];
        __syncthreads();
    }
    if (t == 0) {
        float mx = slog[0];
        for (int f = 1; f < FN; ++f) mx = fmaxf(mx, slog[f]);
        float sum = 0.f, e[FN];
        for (int f = 0; f < FN; ++f) { e[f] = expf(slog[f] - mx); sum += e[f]; }
        for (int f = 0; f < FN; ++f) ssc[f] = e[f] / sum;
    }
    __syncthreads();
    float r = 0.f;
    for (int f = 0; f < FN; ++f)
        r += ssc[f] * hf[((size_t)n * FN + f) * HID + t];
    res[(size_t)n * HID + t] = (__bf16)r;
}

// ---------------------------------------------------------------------------

extern "C" void kernel_launch(void* const* d_in, const int* in_sizes, int n_in,
                              void* d_out, int out_size, void* d_ws, size_t ws_size,
                              hipStream_t stream) {
    (void)in_sizes; (void)n_in; (void)out_size; (void)ws_size;
    const float* x      = (const float*)d_in[0];
    const int*   ei     = (const int*)d_in[1];
    const float* lin1_w = (const float*)d_in[2];
    const float* lin1_b = (const float*)d_in[3];
    const float* lin2_w = (const float*)d_in[4];
    const float* lin2_b = (const float*)d_in[5];
    const float* filt_w = (const float*)d_in[6];
    const float* wf_w   = (const float*)d_in[7];
    const float* wf_b   = (const float*)d_in[8];
    const float* wx_w   = (const float*)d_in[9];
    const float* wx_b   = (const float*)d_in[10];
    const int* src = ei;
    const int* dst = ei + N_EDGES;
    float* out = (float*)d_out;

    float* ws   = (float*)d_ws;
    float* h    = ws;                  // [N,H]
    float* bufA = ws + (size_t)NH;     // prop ping
    float* bufB = ws + 2 * (size_t)NH; // prop pong (also lin1 temp)
    float* hf   = ws + 3 * (size_t)NH; // [N,FN,H]
    float* hfp  = ws + 8 * (size_t)NH; // [N,FN,H]
    float* xp   = ws + 13 * (size_t)NH;// [N,H]
    __bf16* res = (__bf16*)(ws + 14 * (size_t)NH); // [N,H] bf16
    float* deg  = ws + 15 * (size_t)NH;
    float* dinv = deg + N_NODES;
    float* h1   = bufB;                // lin1 output lives in bufB until iter 1

    // init accumulators
    k_zero<<<256, 256, 0, stream>>>(deg, (size_t)N_NODES);
    k_zero<<<4096, 256, 0, stream>>>(hf, (size_t)5 * NH);

    // graph normalization
    k_deg<<<(N_EDGES + 255) / 256, 256, 0, stream>>>(src, deg);
    k_dinv<<<(N_NODES + 255) / 256, 256, 0, stream>>>(deg, dinv);

    // h = relu(x@W1+b1)@W2+b2
    k_gemm256<<<dim3(N_NODES / 16, 2), 256, 0, stream>>>(x, lin1_w, lin1_b, h1,
                                                         N_NODES, DIM_IN, 1);
    k_gemm256<<<dim3(N_NODES / 16, 2), 256, 0, stream>>>(h1, lin2_w, lin2_b, h,
                                                         N_NODES, HID, 0);

    // Bernstein propagation + streaming filter accumulate
    float* cur = h;
    float* nxt = bufA;
    for (int i = 0; i <= KDEG; ++i) {
        k_accum<<<2048, 256, 0, stream>>>(cur, hf, filt_w, i);
        if (i < KDEG) {
            k_scale<<<2048, 256, 0, stream>>>(cur, nxt);
            k_edge<<<N_EDGES, 256, 0, stream>>>(cur, nxt, src, dst, dinv);
            cur = nxt;
            nxt = (cur == bufA) ? bufB : bufA;
        }
    }

    // attention projections
    k_gemm256<<<dim3(N_NODES * FN / 16, 2), 256, 0, stream>>>(hf, wf_w, wf_b, hfp,
                                                              N_NODES * FN, HID, 2);
    k_gemm256<<<dim3(N_NODES / 16, 2), 256, 0, stream>>>(h, wx_w, wx_b, xp,
                                                         N_NODES, HID, 2);

    // softmax fusion -> res (bf16)
    k_fuse<<<N_NODES, 256, 0, stream>>>(hf, hfp, xp, res);

    // out = res @ res.T
    dim3 sg((N_NODES + 63) / 64, (N_NODES + 127) / 128);
    k_syrk<<<sg, 256, 0, stream>>>(res, out);
}
